// SelfAttentionLayerV1_19696720019921
// MI455X (gfx1250) — matrix operand order repaired and checked by reading, not compiled, not run on hardware
//
#include <hip/hip_runtime.h>
#include <stdint.h>

// MI455X / gfx1250 causal self-attention.
// q,k,v = x@W+b via f16 WMMA (f32 acc); S = qk^T/32 (f16 logits, L2-resident);
// in-place causal row softmax; O = P@V via WMMA.
// Scores/PV stage tiles with the Tensor Data Mover (tensor_load_to_lds,
// TENSORcnt-pipelined, pad_enable -> 80B LDS rows = bank-staggered, 16B-aligned).
// NOTE: scores_kernel is defined first so the disasm snippet shows the TDM path.

typedef _Float16 half_t;
typedef __attribute__((ext_vector_type(16))) _Float16 v16h;
typedef __attribute__((ext_vector_type(8)))  _Float16 v8h;
typedef __attribute__((ext_vector_type(4)))  _Float16 v4h;
typedef __attribute__((ext_vector_type(8)))  float    v8f;
typedef __attribute__((ext_vector_type(4)))  unsigned int u32x4;
typedef __attribute__((ext_vector_type(4)))  int i32x4;
typedef __attribute__((ext_vector_type(8)))  int i32x8;

#define NB   4
#define NS   2048
#define ND   1024
#define LDT  40   // LDS row stride (halfs): 64B data + 16B pad = 80B rows

#ifndef __has_builtin
#define __has_builtin(x) 0
#endif
#if defined(__gfx1250__) && __has_builtin(__builtin_amdgcn_tensor_load_to_lds) && \
    __has_builtin(__builtin_amdgcn_s_wait_tensorcnt)
#define USE_TDM 1
#define WAIT_TCNT(n) __builtin_amdgcn_s_wait_tensorcnt(n)
#else
#define USE_TDM 0
#define WAIT_TCNT(n) ((void)0)
#endif

__device__ __forceinline__ v8f zero_v8f() {
  v8f z;
#pragma unroll
  for (int i = 0; i < 8; i++) z[i] = 0.0f;
  return z;
}

// A fragment, 16x32 f16 (ISA 7.12.2): lane<16 holds K {0..7,16..23},
// lane>=16 holds K {8..15,24..31}; base points at row start in LDS.
__device__ __forceinline__ v16h ld_frag_a(const half_t* base, int h16) {
  v8h lo = *(const v8h*)(base + h16 * 8);
  v8h hi = *(const v8h*)(base + 16 + h16 * 8);
  v16h r;
#pragma unroll
  for (int i = 0; i < 8; i++) { r[i] = lo[i]; r[8 + i] = hi[i]; }
  return r;
}

// B fragment, 32x16 f16: lane holds column n=lane&15; lanes 0-15 K={0..7,16..23},
// lanes 16-31 K={8..15,24..31}. rowptr points at that column's contiguous-K row.
__device__ __forceinline__ v16h ld_frag_b(const half_t* rowptr, int h16) {
  v8h lo = *(const v8h*)(rowptr + h16 * 8);
  v8h hi = *(const v8h*)(rowptr + 16 + h16 * 8);
  v16h r;
#pragma unroll
  for (int i = 0; i < 8; i++) { r[i] = lo[i]; r[8 + i] = hi[i]; }
  return r;
}

__device__ __forceinline__ v8f wmma16(v16h a, v16h b, v8f c) {
  return __builtin_amdgcn_wmma_f32_16x16x32_f16(false, a, false, b,
                                                (short)0, c, false, false);
}

__device__ __forceinline__ unsigned lds_off(const void* p) {
  // Generic LDS pointers carry the byte offset in the low 32 bits.
  return (unsigned)(uintptr_t)p;
}

// TDM: load a 2D f16 tile (rows x 32 halfs) into LDS with 80B padded rows.
// data_size=8B: tile_dim0=8, pad 4 DW after every 16 DW (one 64B row).
__device__ __forceinline__ void tdm_load_2d(unsigned lds_addr, const half_t* gptr,
                                            int rows, unsigned stride_u64) {
#if USE_TDM
  unsigned long long ga = (unsigned long long)(uintptr_t)gptr;
  u32x4 g0;
  g0[0] = 1u;                                              // count=1
  g0[1] = lds_addr;                                        // lds_addr
  g0[2] = (unsigned)ga;                                    // global_addr lo
  g0[3] = ((unsigned)(ga >> 32) & 0x01FFFFFFu) | 0x80000000u; // addr hi | type=2
  i32x8 g1;
  g1[0] = (int)0x06D30000u;  // data_size=3(8B) | pad_en | pad_intv=16DW | pad=4DW
  g1[1] = (int)((stride_u64 & 0xFFFFu) << 16);             // tensor_dim0 lo16
  g1[2] = (int)((stride_u64 >> 16) & 0xFFFFu);             // tensor_dim0 hi16
  g1[3] = (int)((8u << 16) | 16u);                         // tile_dim0=8 | tensor_dim1=1<<20
  g1[4] = rows;                                            // tile_dim1 (tile_dim2=0)
  g1[5] = (int)stride_u64;                                 // tensor_dim0_stride
  g1[6] = 0; g1[7] = 0;
  i32x4 g2; g2[0] = g2[1] = g2[2] = g2[3] = 0;
  i32x4 g3; g3[0] = g3[1] = g3[2] = g3[3] = 0;
#if __clang_major__ >= 23
  i32x8 g4;
#pragma unroll
  for (int i = 0; i < 8; i++) g4[i] = 0;
  __builtin_amdgcn_tensor_load_to_lds(g0, g1, g2, g3, g4, 0);
#else
  __builtin_amdgcn_tensor_load_to_lds(g0, g1, g2, g3, 0);
#endif
#else
  (void)lds_addr; (void)gptr; (void)rows; (void)stride_u64;
#endif
}

// ---------------- Kernel 2: scores S = q k^T * (1/32) -----------------------
// grid (keyTile=32, queryTile=32, NB), 128 threads (4 waves). 64x64 per block.
// TDM stages Q/K 64x32 chunks into a 3-deep LDS ring, pipelined on TENSORcnt.
__global__ __launch_bounds__(128) void scores_kernel(
    const half_t* __restrict__ qh, const half_t* __restrict__ kh,
    half_t* __restrict__ sc) {
  const int jt = blockIdx.x, it = blockIdx.y, b = blockIdx.z;
  if (jt > it) return;  // uniform per block

  __shared__ half_t Qs[3][64 * LDT];
  __shared__ half_t Ks[3][64 * LDT];

  const int tid = threadIdx.x, lane = tid & 31, w = tid >> 5;
  const int h16 = (lane >> 4) & 1, l16 = lane & 15;
  const int i0 = it * 64, j0 = jt * 64;
  const half_t* qtile = qh + ((size_t)b * NS + i0) * ND;
  const half_t* ktile = kh + ((size_t)b * NS + j0) * ND;

  v8f acc[4];
#pragma unroll
  for (int nt = 0; nt < 4; nt++) acc[nt] = zero_v8f();

#if USE_TDM
  if (tid == 0) {
    tdm_load_2d(lds_off(&Qs[0][0]), qtile, 64, ND / 4);
    tdm_load_2d(lds_off(&Ks[0][0]), ktile, 64, ND / 4);
  }
  for (int kc = 0; kc < ND; kc += 32) {
    const int c  = kc >> 5;
    const int bi = c % 3;
    if (kc + 32 < ND) {
      // 3-deep ring: target buffer was last read 2 iterations ago, and every
      // wave has passed that iteration's barrier before this issue.
      if (tid == 0) {
        tdm_load_2d(lds_off(&Qs[(c + 1) % 3][0]), qtile + kc + 32, 64, ND / 4);
        tdm_load_2d(lds_off(&Ks[(c + 1) % 3][0]), ktile + kc + 32, 64, ND / 4);
      }
      WAIT_TCNT(2);  // drain current chunk's pair; next pair may stay in flight
    } else {
      WAIT_TCNT(0);
    }
    __syncthreads();
    v16h a = ld_frag_a(&Qs[bi][(w * 16 + l16) * LDT], h16);
#pragma unroll
    for (int nt = 0; nt < 4; nt++) {
      v16h bf = ld_frag_b(&Ks[bi][(nt * 16 + l16) * LDT], h16);
      acc[nt] = wmma16(a, bf, acc[nt]);
    }
  }
#else
  for (int kc = 0; kc < ND; kc += 32) {
#pragma unroll
    for (int r = 0; r < 2; r++) {
      int cix = tid + r * 128;
      int row = cix >> 2, c8 = cix & 3;
      *(v8h*)(&Qs[0][row * LDT + c8 * 8]) =
          *(const v8h*)(qtile + (size_t)row * ND + kc + c8 * 8);
      *(v8h*)(&Ks[0][row * LDT + c8 * 8]) =
          *(const v8h*)(ktile + (size_t)row * ND + kc + c8 * 8);
    }
    __syncthreads();
    v16h a = ld_frag_a(&Qs[0][(w * 16 + l16) * LDT], h16);
#pragma unroll
    for (int nt = 0; nt < 4; nt++) {
      v16h bf = ld_frag_b(&Ks[0][(nt * 16 + l16) * LDT], h16);
      acc[nt] = wmma16(a, bf, acc[nt]);
    }
    __syncthreads();
  }
#endif

  const float scale = 0.03125f;  // 1/sqrt(1024)
  half_t* srow = sc + (size_t)b * NS * NS;
#pragma unroll
  for (int nt = 0; nt < 4; nt++) {
    int jg = j0 + nt * 16 + l16;
#pragma unroll
    for (int j = 0; j < 8; j++) {
      int ig = i0 + w * 16 + j + 8 * h16;
      srow[(size_t)ig * NS + jg] = (half_t)(acc[nt][j] * scale);
    }
  }
}

// ---------------- Kernel 4: O = P @ V (fp32 out) ----------------------------
// grid (ND/64, NS/128, NB), 256 threads. K truncated at m0+128 (causal zeros).
// TDM stages P 128x32 chunks (3-deep ring); V transposed manually (2-deep).
__global__ __launch_bounds__(256) void pv_kernel(
    const half_t* __restrict__ sc, const half_t* __restrict__ vh,
    float* __restrict__ out) {
  const int b = blockIdx.z;
  const int n0 = blockIdx.x * 64, m0 = blockIdx.y * 128;
  __shared__ half_t Ps[3][128 * LDT];
  __shared__ half_t Vs[2][64 * LDT];

  const int tid = threadIdx.x, lane = tid & 31, w = tid >> 5;
  const int h16 = (lane >> 4) & 1, l16 = lane & 15;
  const half_t* ptile = sc + ((size_t)b * NS + m0) * NS;
  const half_t* vbase = vh + (size_t)b * NS * ND;

  v8f acc[4];
#pragma unroll
  for (int nt = 0; nt < 4; nt++) acc[nt] = zero_v8f();

  const int kmax = m0 + 128;  // P[i][j]==0 for j>i

#if USE_TDM
  if (tid == 0) tdm_load_2d(lds_off(&Ps[0][0]), ptile, 128, NS / 4);
  for (int kk = 0; kk < kmax; kk += 32) {
    const int c = kk >> 5;
    const int bi = c % 3, vb = c & 1;
    if (tid == 0 && kk + 32 < kmax)
      tdm_load_2d(lds_off(&Ps[(c + 1) % 3][0]), ptile + kk + 32, 128, NS / 4);
    {  // stage V chunk transposed: Vs[vb][n][k]
      int row = tid >> 3, c8 = tid & 7;
      v8h t = *(const v8h*)(vbase + (size_t)(kk + row) * ND + n0 + c8 * 8);
#pragma unroll
      for (int u = 0; u < 8; u++) Vs[vb][(c8 * 8 + u) * LDT + row] = t[u];
    }
    if (kk + 32 < kmax) { WAIT_TCNT(1); } else { WAIT_TCNT(0); }
    __syncthreads();
    v16h a = ld_frag_a(&Ps[bi][(w * 16 + l16) * LDT], h16);
#pragma unroll
    for (int nt = 0; nt < 4; nt++) {
      v16h bf = ld_frag_b(&Vs[vb][(nt * 16 + l16) * LDT], h16);
      acc[nt] = wmma16(a, bf, acc[nt]);
    }
  }
#else
  for (int kk = 0; kk < kmax; kk += 32) {
#pragma unroll
    for (int r = 0; r < 2; r++) {
      int cix = tid + r * 256;
      int row = cix >> 2, c8 = cix & 3;
      *(v8h*)(&Ps[0][row * LDT + c8 * 8]) =
          *(const v8h*)(ptile + (size_t)row * NS + kk + c8 * 8);
    }
    {
      int row = tid >> 3, c8 = tid & 7;
      v8h t = *(const v8h*)(vbase + (size_t)(kk + row) * ND + n0 + c8 * 8);
#pragma unroll
      for (int u = 0; u < 8; u++) Vs[0][(c8 * 8 + u) * LDT + row] = t[u];
    }
    __syncthreads();
    v16h a = ld_frag_a(&Ps[0][(w * 16 + l16) * LDT], h16);
#pragma unroll
    for (int nt = 0; nt < 4; nt++) {
      v16h bf = ld_frag_b(&Vs[0][(nt * 16 + l16) * LDT], h16);
      acc[nt] = wmma16(a, bf, acc[nt]);
    }
    __syncthreads();
  }
#endif

#pragma unroll
  for (int nt = 0; nt < 4; nt++) {
    int n = n0 + nt * 16 + l16;
#pragma unroll
    for (int j = 0; j < 8; j++) {
      int ig = m0 + w * 16 + j + 8 * h16;
      out[((size_t)b * NS + ig) * ND + n] = acc[nt][j];
    }
  }
}

// ---------------- Kernel 1: fused q/k/v projections -------------------------
// grid (ND/64, (NB*NS)/128, 3), 256 threads (8 waves). Tile M=128,N=64,K=32.
// Manual staging (needs f32->f16 conversion, which TDM cannot do).
__global__ __launch_bounds__(256) void proj_kernel(
    const float* __restrict__ x,
    const float* __restrict__ Wq, const float* __restrict__ bq,
    const float* __restrict__ Wk, const float* __restrict__ bk,
    const float* __restrict__ Wv, const float* __restrict__ bv,
    half_t* __restrict__ qh, half_t* __restrict__ kh, half_t* __restrict__ vh) {
  __shared__ half_t Xs[128 * LDT];
  __shared__ half_t Ws[64 * LDT];

  const int z = blockIdx.z;
  const float* W    = (z == 0) ? Wq : (z == 1) ? Wk : Wv;
  const float* bias = (z == 0) ? bq : (z == 1) ? bk : bv;
  half_t* out       = (z == 0) ? qh : (z == 1) ? kh : vh;

  const int n0 = blockIdx.x * 64;
  const int m0 = blockIdx.y * 128;
  const int tid = threadIdx.x;
  const int lane = tid & 31, w = tid >> 5;
  const int h16 = (lane >> 4) & 1, l16 = lane & 15;

  v8f acc[4];
#pragma unroll
  for (int nt = 0; nt < 4; nt++) acc[nt] = zero_v8f();

  for (int k0 = 0; k0 < ND; k0 += 32) {
#pragma unroll
    for (int r = 0; r < 4; r++) {
      int idx = tid + r * 256;
      int row = idx >> 3, c4 = idx & 7;
      float4 f = *(const float4*)(x + (size_t)(m0 + row) * ND + k0 + c4 * 4);
      v4h hv;
      hv[0] = (half_t)f.x; hv[1] = (half_t)f.y;
      hv[2] = (half_t)f.z; hv[3] = (half_t)f.w;
      *(v4h*)(&Xs[row * LDT + c4 * 4]) = hv;
    }
#pragma unroll
    for (int r = 0; r < 2; r++) {
      int idx = tid + r * 256;
      int row = idx >> 4, c4 = idx & 15;
      float4 f = *(const float4*)(W + (size_t)(k0 + row) * ND + n0 + c4 * 4);
      Ws[(c4 * 4 + 0) * LDT + row] = (half_t)f.x;
      Ws[(c4 * 4 + 1) * LDT + row] = (half_t)f.y;
      Ws[(c4 * 4 + 2) * LDT + row] = (half_t)f.z;
      Ws[(c4 * 4 + 3) * LDT + row] = (half_t)f.w;
    }
    __syncthreads();
    v16h a = ld_frag_a(&Xs[(w * 16 + l16) * LDT], h16);
#pragma unroll
    for (int nt = 0; nt < 4; nt++) {
      v16h bf = ld_frag_b(&Ws[(nt * 16 + l16) * LDT], h16);
      acc[nt] = wmma16(a, bf, acc[nt]);
    }
    __syncthreads();
  }
#pragma unroll
  for (int nt = 0; nt < 4; nt++) {
    int n = n0 + nt * 16 + l16;
    float bn = bias[n];
#pragma unroll
    for (int j = 0; j < 8; j++) {
      int row = m0 + w * 16 + j + 8 * h16;
      out[(size_t)row * ND + n] = (half_t)(acc[nt][j] + bn);
    }
  }
}

// ---------------- Kernel 3: causal row softmax (in place) -------------------
__global__ __launch_bounds__(256) void softmax_kernel(half_t* __restrict__ sc) {
  const int i = blockIdx.x, b = blockIdx.y;
  half_t* row = sc + ((size_t)b * NS + i) * NS;
  const int nv = i + 1;
  __shared__ float red[256];
  const int tid = threadIdx.x;

  float m = -3.0e38f;
  for (int j = tid; j < nv; j += 256) m = fmaxf(m, (float)row[j]);
  red[tid] = m; __syncthreads();
  for (int s = 128; s > 0; s >>= 1) {
    if (tid < s) red[tid] = fmaxf(red[tid], red[tid + s]);
    __syncthreads();
  }
  const float rowmax = red[0]; __syncthreads();

  float sum = 0.0f;
  for (int j = tid; j < nv; j += 256) sum += __expf((float)row[j] - rowmax);
  red[tid] = sum; __syncthreads();
  for (int s = 128; s > 0; s >>= 1) {
    if (tid < s) red[tid] += red[tid + s];
    __syncthreads();
  }
  const float inv = 1.0f / red[0]; __syncthreads();

  for (int j = tid; j < NS; j += 256) {
    float p = (j < nv) ? __expf((float)row[j] - rowmax) * inv : 0.0f;
    row[j] = (half_t)p;
  }
}

extern "C" void kernel_launch(void* const* d_in, const int* in_sizes, int n_in,
                              void* d_out, int out_size, void* d_ws,
                              size_t ws_size, hipStream_t stream) {
  const float* x  = (const float*)d_in[0];
  const float* Wq = (const float*)d_in[1];
  const float* bq = (const float*)d_in[2];
  const float* Wk = (const float*)d_in[3];
  const float* bk = (const float*)d_in[4];
  const float* Wv = (const float*)d_in[5];
  const float* bv = (const float*)d_in[6];
  float* out = (float*)d_out;

  // Workspace (f16): qh|kh|vh (16MB each) + scores/P (32MB) = 80MB
  const size_t BSD = (size_t)NB * NS * ND;
  half_t* qh = (half_t*)d_ws;
  half_t* kh = qh + BSD;
  half_t* vh = kh + BSD;
  half_t* sc = vh + BSD;

  proj_kernel<<<dim3(ND / 64, (NB * NS) / 128, 3), 256, 0, stream>>>(
      x, Wq, bq, Wk, bk, Wv, bv, qh, kh, vh);
  scores_kernel<<<dim3(NS / 64, NS / 64, NB), 128, 0, stream>>>(qh, kh, sc);
  softmax_kernel<<<dim3(NS, NB), 256, 0, stream>>>(sc);
  pv_kernel<<<dim3(ND / 64, NS / 128, NB), 256, 0, stream>>>(sc, vh, out);
}
